// MultiHeadCrossAttention_49727131353392
// MI455X (gfx1250) — compile-verified, matrix operand order checked by reading, not run on hardware
//
#include <hip/hip_runtime.h>
#include <math.h>

// ---------------- problem constants (from reference) ----------------
#define B_  32
#define C_  128
#define H_  8
#define NC_ 2048          // concat length
#define CHUNK_ROWS 64     // comp rows per chunk  -> 512 kv positions
#define NCHUNK (NC_ / CHUNK_ROWS)       // 32
#define SCALE 0.25f       // hd^-0.5, hd=16

typedef __attribute__((ext_vector_type(16))) __bf16 v16bf;
typedef __attribute__((ext_vector_type(8)))  float  v8f;

union FragU { v16bf v; uint4 q[2]; };

__device__ __forceinline__ unsigned short f2bf(float f) {
  union { float f; unsigned u; } v; v.f = f;
  unsigned r = v.u + 0x7FFFu + ((v.u >> 16) & 1u);   // RNE
  return (unsigned short)(r >> 16);
}

// Load one 16x32 bf16 WMMA fragment half-pair. Row-major [row][K], stride in
// elements. k0 must be a multiple of 8 (16B aligned) -> two b128 loads.
__device__ __forceinline__ v16bf load_frag(const unsigned short* base, int row,
                                           int stride, int k0) {
  FragU f;
  const unsigned short* p = base + row * stride + k0;
  f.q[0] = *(const uint4*)(p);
  f.q[1] = *(const uint4*)(p + 16);
  return f.v;
}

#define TK_SWAP(a, b) { float _tt = a; a = b; b = _tt; }
#define TK_INSERT(vv) do { float _v = (vv); if (_v > t11) { t11 = _v;          \
  if (t11 > t10) TK_SWAP(t10, t11);                                            \
  if (t10 > t9 ) TK_SWAP(t9 , t10);                                            \
  if (t9  > t8 ) TK_SWAP(t8 , t9 );                                            \
  if (t8  > t7 ) TK_SWAP(t7 , t8 );                                            \
  if (t7  > t6 ) TK_SWAP(t6 , t7 );                                            \
  if (t6  > t5 ) TK_SWAP(t5 , t6 );                                            \
  if (t5  > t4 ) TK_SWAP(t4 , t5 );                                            \
  if (t4  > t3 ) TK_SWAP(t3 , t4 );                                            \
  if (t3  > t2 ) TK_SWAP(t2 , t3 );                                            \
  if (t2  > t1 ) TK_SWAP(t1 , t2 );                                            \
  if (t1  > t0 ) TK_SWAP(t0 , t1 ); } } while (0)

#define TK_RESET() { t0=t1=t2=t3=t4=t5=t6=t7=t8=t9=t10=t11=-3.402823466e38f; }

// ---------------- Tensor Data Mover (TDM) tile load ----------------
#if __has_builtin(__builtin_amdgcn_tensor_load_to_lds)
#define USE_TDM 1
typedef unsigned int tdm_v4u __attribute__((ext_vector_type(4)));
typedef int tdm_v8i __attribute__((ext_vector_type(8)));
typedef int tdm_v4i __attribute__((ext_vector_type(4)));

// DMA a 64x128 bf16 row-major tile from global `gsrc` into LDS offset 0.
// D# group0: count=1, lds_addr=0, global_addr, type=2 ("image").
// D# group1: data_size=1 (2B), tensor_dim0=128, tensor_dim1=64,
//            tile_dim0=128, tile_dim1=64, tensor_dim0_stride=128.
__device__ __forceinline__ void tdm_load_tile(const unsigned short* gsrc) {
  unsigned long long ga = (unsigned long long)(const void*)gsrc;
  tdm_v4u g0;
  g0.x = 1u;                                         // count=1, user mode
  g0.y = 0u;                                         // lds_addr = 0 (compA)
  g0.z = (unsigned)(ga & 0xFFFFFFFFu);               // global_addr[31:0]
  g0.w = (unsigned)((ga >> 32) & 0x01FFFFFFu) | 0x80000000u;  // [56:32]|type=2
  tdm_v8i g1;
  g1[0] = 0x00010000;        // wg_mask=0, data_size=1 (2 bytes)
  g1[1] = 128 << 16;         // tensor_dim0 = 128 (elements)
  g1[2] = 64 << 16;          // tensor_dim1 = 64
  g1[3] = 128 << 16;         // tile_dim0 = 128
  g1[4] = 64;                // tile_dim1 = 64, tile_dim2 = 0
  g1[5] = 128;               // tensor_dim0_stride = 128
  g1[6] = 0;
  g1[7] = 0;                 // tensor_dim1_stride = 0 (2D)
  tdm_v4i z4 = {0, 0, 0, 0};
#if __clang_major__ >= 23
  tdm_v8i z8 = {0, 0, 0, 0, 0, 0, 0, 0};
  __builtin_amdgcn_tensor_load_to_lds(g0, g1, z4, z4, z8, 0);
#else
  __builtin_amdgcn_tensor_load_to_lds(g0, g1, z4, z4, 0);
#endif
}
#endif

// ------------- kernel 0: comp = concat(x, complement) as bf16 -------------
__global__ __launch_bounds__(256) void k_comp(const float* __restrict__ x,
                                              const float* __restrict__ complement,
                                              unsigned short* __restrict__ compBF) {
  size_t i0 = ((size_t)blockIdx.x * 256 + threadIdx.x) * 4;
#pragma unroll
  for (int k = 0; k < 4; ++k) {
    size_t idx = i0 + k;                   // over 32*2048*128 = 8388608
    int b = (int)(idx >> 18);
    int rem = (int)(idx & 262143);
    int n = rem >> 7, c = rem & 127;
    float v = (n == 0) ? x[b * 128 + c]
                       : complement[((size_t)b * 2047 + (n - 1)) * 128 + c];
    compBF[idx] = f2bf(v);
  }
}

// ---------------- kernel 1: Q projection  q = x @ Wq ----------------
__global__ __launch_bounds__(256) void k_qproj(const float* __restrict__ x,
                                               const float* __restrict__ Wq,
                                               float* __restrict__ qproj) {
  __shared__ float xs[C_];
  int b = blockIdx.x, tid = threadIdx.x;
  if (tid < C_) xs[tid] = x[b * C_ + tid];
  __syncthreads();
  int col0 = tid * 8;
  float acc[8];
#pragma unroll
  for (int k = 0; k < 8; ++k) acc[k] = 0.f;
  for (int c = 0; c < C_; ++c) {
    float xv = xs[c];
    const float* wr = Wq + c * 2048 + col0;
#pragma unroll
    for (int k = 0; k < 8; ++k) acc[k] += xv * wr[k];
  }
#pragma unroll
  for (int k = 0; k < 8; ++k) qproj[b * 2048 + col0 + k] = acc[k];
}

// ------- kernel 2: fold Q into K-side weights, per (b,h), transposed -------
// qwT[(b,h)][j][c] = sum_d Q[b,h,q,d] * Wkv[c][e*256 + h*16 + d],  j = q*8+e
__global__ __launch_bounds__(256) void k_foldqw(const float* __restrict__ qproj,
                                                const float* __restrict__ Wkv,
                                                unsigned short* __restrict__ qwT) {
  int wg = blockIdx.x;               // b*8 + h
  int b = wg >> 3, h = wg & 7;
  unsigned short* out = qwT + (size_t)wg * (128 * 128);
  for (int idx = threadIdx.x; idx < 128 * 128; idx += blockDim.x) {
    int j = idx >> 7, c = idx & 127;
    int q = j >> 3, e = j & 7;
    const float* qp = qproj + b * 2048 + q * 128 + h * 16;
    const float* wk = Wkv + c * 2048 + e * 256 + h * 16;
    float s = 0.f;
#pragma unroll
    for (int d = 0; d < 16; ++d) s += qp[d] * wk[d];
    out[idx] = f2bf(s);
  }
}

// ------- kernel 3: V-side head-sliced weights, transposed, per h -------
// wvT[h][ed][c] = Wkv[c][e*256 + 128 + h*16 + d],  ed = e*16 + d
__global__ __launch_bounds__(256) void k_foldwv(const float* __restrict__ Wkv,
                                                unsigned short* __restrict__ wvT) {
  int h = blockIdx.x;
  unsigned short* out = wvT + (size_t)h * (128 * 128);
  for (int idx = threadIdx.x; idx < 128 * 128; idx += blockDim.x) {
    int ed = idx >> 7, c = idx & 127;
    int e = ed >> 4, d = ed & 15;
    out[idx] = f2bf(Wkv[c * 2048 + e * 256 + 128 + h * 16 + d]);
  }
}

// ---------------- kernel 4: fused attention per (b,h) ----------------
__global__ __launch_bounds__(256) void k_attn(
    const unsigned short* __restrict__ compBF,
    const unsigned short* __restrict__ qwT_all,
    const unsigned short* __restrict__ wvT_all, float* __restrict__ o) {
  // 64 KB static LDS, regions overlaid in time:
  //  [0,16K)   compA  bf16 [64][128]   (TDM dest; reduction buffer at the end)
  //  [16K,48K) Sbuf   f32  [64][128]  (pass1) / Pbuf bf16 [16][512] (pass2)
  //            mergeBuf f32 [16*16*12] (pass1 end)
  //  [32K,+192) thr/rmax/denom (written only after last Sbuf use)
  //  [48K,64K) Vt    bf16 [16][512]
  __shared__ __align__(16) unsigned char smem[65536];
  unsigned short* compA   = (unsigned short*)smem;
  float*          Sbuf    = (float*)(smem + 16384);
  unsigned short* Pbuf    = (unsigned short*)(smem + 16384);
  float*          mergeBuf= (float*)(smem + 16384);
  float*          thr     = (float*)(smem + 32768);
  float*          rmax    = thr + 16;
  float*          denom   = thr + 32;
  unsigned short* Vt      = (unsigned short*)(smem + 49152);
  float*          redbuf  = (float*)smem;

  int wg = blockIdx.x;
  int b = wg >> 3, h = wg & 7;
  int tid = threadIdx.x;
  int lane = tid & 31, wave = tid >> 5;
  int l15 = lane & 15, half = lane >> 4;
  const unsigned short* qwT = qwT_all + (size_t)wg * (128 * 128);
  const unsigned short* wvT = wvT_all + (size_t)h * (128 * 128);
  const unsigned short* compB = compBF + (size_t)b * (NC_ * 128);

  float t0, t1, t2, t3, t4, t5, t6, t7, t8, t9, t10, t11;
  TK_RESET();

  // ================= pass 1: raw scores -> per-row top-12 =================
  for (int ch = 0; ch < NCHUNK; ++ch) {
    if (ch + 1 < NCHUNK)
      __builtin_prefetch(compB + (size_t)(ch + 1) * CHUNK_ROWS * 128 + tid * 8, 0, 0);
    // stage comp chunk (64x128 bf16) into LDS
#ifdef USE_TDM
    if (wave == 0) {
      tdm_load_tile(compB + (size_t)ch * CHUNK_ROWS * 128);
      __builtin_amdgcn_s_wait_tensorcnt(0);
    }
#else
    {
      const uint4* src = (const uint4*)(compB + (size_t)ch * CHUNK_ROWS * 128);
      uint4* dst = (uint4*)compA;
      for (int i = tid; i < 1024; i += 256) dst[i] = src[i];
    }
#endif
    __syncthreads();
    // S[n][j] = compA @ qwT^T  : 4 Mtiles x 8 Ntiles, K=128 (4 wmma)
#pragma unroll
    for (int s = 0; s < 4; ++s) {
      int t = wave * 4 + s;
      int m = t >> 3, jt = t & 7;
      v8f acc = {0.f, 0.f, 0.f, 0.f, 0.f, 0.f, 0.f, 0.f};
#pragma unroll
      for (int kk = 0; kk < 4; ++kk) {
        v16bf a  = load_frag(compA, m * 16 + l15, 128, kk * 32 + half * 8);
        v16bf bb = load_frag(qwT,  jt * 16 + l15, 128, kk * 32 + half * 8);
        acc = __builtin_amdgcn_wmma_f32_16x16x32_bf16(false, a, false, bb,
                                                      (short)0, acc, false, false);
      }
      int j = jt * 16 + l15;
#pragma unroll
      for (int r = 0; r < 8; ++r) {
        int n = m * 16 + r + 8 * half;
        Sbuf[n * 128 + j] = acc[r];
      }
    }
    __syncthreads();
    // per-row streaming top-12: thread (q,w) scans rows n=w*4..w*4+3
    {
      int q = tid >> 4, w = tid & 15;
#pragma unroll
      for (int nn = 0; nn < 4; ++nn) {
        int n = w * 4 + nn;
#pragma unroll
        for (int e = 0; e < 8; ++e) {
          float v = Sbuf[n * 128 + q * 8 + e];
          TK_INSERT(v);
        }
      }
    }
    __syncthreads();
  }
  // merge 16 partial top-12 lists per row
  {
    int q = tid >> 4, w = tid & 15;
    float* mb = mergeBuf + (q * 16 + w) * 12;
    mb[0] = t0;  mb[1] = t1;  mb[2] = t2;   mb[3] = t3;
    mb[4] = t4;  mb[5] = t5;  mb[6] = t6;   mb[7] = t7;
    mb[8] = t8;  mb[9] = t9;  mb[10] = t10; mb[11] = t11;
  }
  __syncthreads();
  if (tid < 16) {
    TK_RESET();
    const float* mb = mergeBuf + tid * 192;
    for (int i = 0; i < 192; ++i) { float v = mb[i]; TK_INSERT(v); }
    thr[tid] = t11;     // raw-score threshold (scale > 0, order-preserving)
    rmax[tid] = t0;     // raw-score row max
    denom[tid] = 0.f;
  }
  __syncthreads();

  // ============ pass 2: recompute scores -> P, V, O += P @ V ============
  v8f co = {0.f, 0.f, 0.f, 0.f, 0.f, 0.f, 0.f, 0.f};
  for (int ch = 0; ch < NCHUNK; ++ch) {
    if (ch + 1 < NCHUNK)
      __builtin_prefetch(compB + (size_t)(ch + 1) * CHUNK_ROWS * 128 + tid * 8, 0, 0);
#ifdef USE_TDM
    if (wave == 0) {
      tdm_load_tile(compB + (size_t)ch * CHUNK_ROWS * 128);
      __builtin_amdgcn_s_wait_tensorcnt(0);
    }
#else
    {
      const uint4* src = (const uint4*)(compB + (size_t)ch * CHUNK_ROWS * 128);
      uint4* dst = (uint4*)compA;
      for (int i = tid; i < 1024; i += 256) dst[i] = src[i];
    }
#endif
    __syncthreads();
    // scores -> thresholded exp -> Pbuf[q][kv] + denom accumulation
#pragma unroll
    for (int s = 0; s < 4; ++s) {
      int t = wave * 4 + s;
      int m = t >> 3, jt = t & 7;
      v8f acc = {0.f, 0.f, 0.f, 0.f, 0.f, 0.f, 0.f, 0.f};
#pragma unroll
      for (int kk = 0; kk < 4; ++kk) {
        v16bf a  = load_frag(compA, m * 16 + l15, 128, kk * 32 + half * 8);
        v16bf bb = load_frag(qwT,  jt * 16 + l15, 128, kk * 32 + half * 8);
        acc = __builtin_amdgcn_wmma_f32_16x16x32_bf16(false, a, false, bb,
                                                      (short)0, acc, false, false);
      }
      int j = jt * 16 + l15;
      int q = j >> 3, e = j & 7;
      float th = thr[q], rm = rmax[q];
      float psum = 0.f;
#pragma unroll
      for (int r = 0; r < 8; ++r) {
        int n = m * 16 + r + 8 * half;
        float sv = acc[r];
        float p = (sv >= th) ? __expf((sv - rm) * SCALE) : 0.f;
        psum += p;
        Pbuf[q * 512 + n * 8 + e] = f2bf(p);
      }
      atomicAdd(&denom[q], psum);       // ds_add_f32
    }
    // V chunk: Vt[d][kv] = (compA @ wvT^T) scattered transposed
#pragma unroll
    for (int s = 0; s < 4; ++s) {
      int t = wave * 4 + s;
      int m = t >> 3, edt = t & 7;
      v8f acc = {0.f, 0.f, 0.f, 0.f, 0.f, 0.f, 0.f, 0.f};
#pragma unroll
      for (int kk = 0; kk < 4; ++kk) {
        v16bf a  = load_frag(compA, m * 16 + l15, 128, kk * 32 + half * 8);
        v16bf bb = load_frag(wvT, edt * 16 + l15, 128, kk * 32 + half * 8);
        acc = __builtin_amdgcn_wmma_f32_16x16x32_bf16(false, a, false, bb,
                                                      (short)0, acc, false, false);
      }
      int ed = edt * 16 + l15;
      int e = ed >> 4, d = ed & 15;
#pragma unroll
      for (int r = 0; r < 8; ++r) {
        int n = m * 16 + r + 8 * half;
        Vt[d * 512 + n * 8 + e] = f2bf(acc[r]);
      }
    }
    __syncthreads();
    // O[q][d] += P[q][kv] * V[kv][d] ; K=512 split as 2 wmma per wave
#pragma unroll
    for (int s2 = 0; s2 < 2; ++s2) {
      int kk = wave * 2 + s2;
      v16bf a  = load_frag(Pbuf, l15, 512, kk * 32 + half * 8);
      v16bf bb = load_frag(Vt,   l15, 512, kk * 32 + half * 8);
      co = __builtin_amdgcn_wmma_f32_16x16x32_bf16(false, a, false, bb,
                                                   (short)0, co, false, false);
    }
    __syncthreads();
  }
  // cross-wave reduction of O partials, normalize, emit o[b][h*256+q*16+d]
#pragma unroll
  for (int r = 0; r < 8; ++r) redbuf[wave * 256 + r * 32 + lane] = co[r];
  __syncthreads();
  {
    float v = 0.f;
#pragma unroll
    for (int w = 0; w < 8; ++w) v += redbuf[w * 256 + tid];
    int r = tid >> 5, ln = tid & 31;
    int q = r + 8 * (ln >> 4);
    int d = ln & 15;
    o[(size_t)b * 2048 + h * 256 + q * 16 + d] = v / denom[q];
  }
}

// ---------------- kernel 5: epilogue  ((o@Wjw)+x)@Wp + bp ----------------
__global__ __launch_bounds__(128) void k_out(const float* __restrict__ o,
                                             const float* __restrict__ x,
                                             const float* __restrict__ Wjw,
                                             const float* __restrict__ Wp,
                                             const float* __restrict__ bp,
                                             float* __restrict__ out) {
  __shared__ float tbuf[128];
  int b = blockIdx.x, c = threadIdx.x;
  float s = x[b * 128 + c];
  const float* ob = o + (size_t)b * 2048;
  for (int j = 0; j < 2048; ++j) s += ob[j] * Wjw[j * 128 + c];
  tbuf[c] = s;
  __syncthreads();
  float s2 = bp[c];
  for (int k = 0; k < 128; ++k) s2 += tbuf[k] * Wp[k * 128 + c];
  out[b * 128 + c] = s2;
}

// ---------------------------- launcher ----------------------------
extern "C" void kernel_launch(void* const* d_in, const int* in_sizes, int n_in,
                              void* d_out, int out_size, void* d_ws, size_t ws_size,
                              hipStream_t stream) {
  (void)in_sizes; (void)n_in; (void)out_size; (void)ws_size;
  const float* x    = (const float*)d_in[0];
  const float* comp = (const float*)d_in[1];   // complement [32,2047,128]
  const float* Wq   = (const float*)d_in[2];
  const float* Wkv  = (const float*)d_in[3];
  const float* Wjw  = (const float*)d_in[4];
  const float* Wp   = (const float*)d_in[5];
  const float* bp   = (const float*)d_in[6];
  float* out = (float*)d_out;
  unsigned char* ws = (unsigned char*)d_ws;

  // workspace layout (~26 MB):
  //   qproj 256KB | o 256KB | qwT 8MB | wvT 256KB | compBF 16MB
  float*          qproj  = (float*)(ws);
  float*          o      = (float*)(ws + 262144);
  unsigned short* qwT    = (unsigned short*)(ws + 524288);
  unsigned short* wvT    = (unsigned short*)(ws + 524288 + 8388608);
  unsigned short* compBF = (unsigned short*)(ws + 524288 + 8388608 + 262144);

  k_comp  <<<8192,      256, 0, stream>>>(x, comp, compBF);
  k_qproj <<<B_,        256, 0, stream>>>(x, Wq, qproj);
  k_foldqw<<<B_ * H_,   256, 0, stream>>>(qproj, Wkv, qwT);
  k_foldwv<<<H_,        256, 0, stream>>>(Wkv, wvT);
  k_attn  <<<B_ * H_,   256, 0, stream>>>(compBF, qwT, wvT, o);
  k_out   <<<B_,        128, 0, stream>>>(o, x, Wjw, Wp, bp, out);
}
